// Attention_layer_32762010534075
// MI455X (gfx1250) — matrix operand order repaired and checked by reading, not compiled, not run on hardware
//
#include <hip/hip_runtime.h>
#include <hip/hip_bf16.h>

// ---------------------------------------------------------------------------
// Attention_layer: B=8, CH=64, D=32, H=64, W=64, S=32
// Per position n=(b,h,w): X (64x32) -> K,Q,V (32x32) -> A=softmax(K^T Q/sqrt(32))
// over first index -> O = V*A -> Out = Wo*O + bo -> y = x + f*Out -> mean over D.
// One wave32 per position; 8 positions (consecutive hw) per 256-thread block.
// All GEMMs via v_wmma_f32_16x16x32_f16 with LDS-staged f16 operands laid out
// to match the CDNA5 WMMA VGPR striping exactly.
// ---------------------------------------------------------------------------

typedef __attribute__((ext_vector_type(16))) _Float16 v16h;
typedef __attribute__((ext_vector_type(8)))  float    v8f;
typedef __attribute__((ext_vector_type(4)))  float    f32x4;

#define CH   64
#define DD   32
#define SS   32
#define HW   4096      // H*W
#define NT   8         // positions per block
#define LDX  64        // ld of Xt [d][c] (f16 elems)
#define LDT  40        // ld of transposed 32x32 f16 mats (pad vs bank conflicts)
#define LDS_SC 33      // ld of f32 score matrix

// per-n LDS block layout (bytes)
#define OFF_XT   0                       // Xt  [32][64] f16   = 4096
#define OFF_KT   4096                    // kT  [32][40] f16   = 2560   (k^T: [d][s])
#define OFF_QT   (4096+2560)             // qT  [32][40] f16
#define OFF_V    (4096+2*2560)           // v   [32][40] f16   ([s][d])
#define OFF_AT   (4096+3*2560)           // aT  [32][40] f16   ([j][i])
#define OFF_OT   (4096+4*2560)           // oT  [32][40] f16   ([d][s])
#define OFF_SC   (4096+5*2560)           // sc  [32][33] f32   = 4224
#define OFF_ORED (OFF_SC+4224)           // outred [64] f32    = 256
#define PER_N    (OFF_ORED+256)          // 21376 bytes (16B aligned)

#define OFF_WK   (PER_N*NT)              // Wk f16 [32][64] 4096
#define OFF_WQ   (OFF_WK+4096)
#define OFF_WV   (OFF_WQ+4096)
#define OFF_WO   (OFF_WV+4096)           // Wo f16 [64][32] 4096
#define OFF_BK   (OFF_WO+4096)           // bk f32[32]
#define OFF_BQ   (OFF_BK+128)
#define OFF_BV   (OFF_BQ+128)
#define OFF_BO   (OFF_BV+128)            // bo f32[64]
#define OFF_XSUM (OFF_BO+256)            // xsum f32 [NT][64] = 2048
#define SMEM_TOTAL (OFF_XSUM+2048)       // 190080 bytes

union V16U { v16h h; f32x4 f[2]; };

// A operand (16x32 f16 tile, M x K). Source: row-major [M][K] LDS mat, ld elems.
// Lanes 0-15: M=lane, halves 0..7 = K kb..kb+7, halves 8..15 = K kb+16..kb+23,
// kb = (lane>=16)*8.  Two contiguous 16-byte LDS loads per lane.
__device__ inline v16h load_a(const _Float16* m0, int ld, int lane) {
  int mrow = lane & 15;
  int kb   = (lane >> 4) << 3;
  const _Float16* p = m0 + mrow * ld + kb;
  V16U u;
  u.f[0] = *(const f32x4*)(p);
  u.f[1] = *(const f32x4*)(p + 16);
  return u.h;
}

// B operand (32x16 f16 tile, K x N). Source: *transposed* row-major [N][K] LDS
// mat (element (k,n) at m0[n*ld + k]).  N=lane&15, halves 0..7 = K kb..kb+7,
// halves 8..15 = K kb+16..kb+23, kb = (lane>=16)*8.  Two 16-byte loads.
__device__ inline v16h load_b(const _Float16* m0, int ld, int lane) {
  int ncol = lane & 15;
  int kb   = (lane >> 4) << 3;
  const _Float16* p = m0 + ncol * ld + kb;
  V16U u;
  u.f[0] = *(const f32x4*)(p);
  u.f[1] = *(const f32x4*)(p + 16);
  return u.h;
}

__device__ inline v8f wmma_f16(v16h a, v16h b, v8f c) {
  return __builtin_amdgcn_wmma_f32_16x16x32_f16(false, a, false, b,
                                                (short)0, c, false, false);
}

// store 16x16 f32 C tile transposed as f16: dst[n][m] (ld elems)
__device__ inline void store_c_t_f16(_Float16* dst, int ld, v8f c, int lane) {
  int n  = lane & 15;
  int mb = (lane >> 4) << 3;
#pragma unroll
  for (int r = 0; r < 8; ++r) dst[n * ld + mb + r] = (_Float16)c[r];
}

// store 16x16 f32 C tile row-major f16: dst[m][n]
__device__ inline void store_c_f16(_Float16* dst, int ld, v8f c, int lane) {
  int n  = lane & 15;
  int mb = (lane >> 4) << 3;
#pragma unroll
  for (int r = 0; r < 8; ++r) dst[(mb + r) * ld + n] = (_Float16)c[r];
}

// projection: dstT[d][s] (or v: dst[s][d]) = W(32x64) @ Xt-staged X + bias
__device__ inline void project(const _Float16* Wl, const float* bias,
                               const _Float16* Xt, _Float16* dstT,
                               bool transpose_out, int lane) {
  int mb = (lane >> 4) << 3;
#pragma unroll
  for (int sb = 0; sb < 2; ++sb) {
    v16h a0 = load_a(Wl + sb * 16 * LDX +  0, LDX, lane);
    v16h a1 = load_a(Wl + sb * 16 * LDX + 32, LDX, lane);
#pragma unroll
    for (int db = 0; db < 2; ++db) {
      v8f c;
#pragma unroll
      for (int r = 0; r < 8; ++r) c[r] = bias[sb * 16 + mb + r];
      v16h b0 = load_b(Xt + db * 16 * LDX +  0, LDX, lane);
      v16h b1 = load_b(Xt + db * 16 * LDX + 32, LDX, lane);
      c = wmma_f16(a0, b0, c);
      c = wmma_f16(a1, b1, c);
      if (transpose_out)
        store_c_t_f16(dstT + db * 16 * LDT + sb * 16, LDT, c, lane);
      else
        store_c_f16(dstT + sb * 16 * LDT + db * 16, LDT, c, lane);
    }
  }
}

__global__ __launch_bounds__(256)
void attn_kernel(const float* __restrict__ x,
                 const float* __restrict__ Wk, const float* __restrict__ bk,
                 const float* __restrict__ Wq, const float* __restrict__ bq,
                 const float* __restrict__ Wv, const float* __restrict__ bv,
                 const float* __restrict__ Wo, const float* __restrict__ bo,
                 const float* __restrict__ factor,
                 float* __restrict__ out) {
  extern __shared__ char smem[];
  const int t    = threadIdx.x;
  const int bid  = blockIdx.x;
  const int b    = bid >> 9;            // 512 hw-chunks per batch
  const int hw0  = (bid & 511) << 3;    // 8 consecutive hw positions

  _Float16* wk_l = (_Float16*)(smem + OFF_WK);
  _Float16* wq_l = (_Float16*)(smem + OFF_WQ);
  _Float16* wv_l = (_Float16*)(smem + OFF_WV);
  _Float16* wo_l = (_Float16*)(smem + OFF_WO);
  float* bk_l = (float*)(smem + OFF_BK);
  float* bq_l = (float*)(smem + OFF_BQ);
  float* bv_l = (float*)(smem + OFF_BV);
  float* bo_l = (float*)(smem + OFF_BO);
  float* xsum = (float*)(smem + OFF_XSUM);

  // ---- cooperative staging: weights -> f16 LDS --------------------------
#pragma unroll
  for (int i = t; i < SS * CH; i += 256) {
    wk_l[i] = (_Float16)Wk[i];
    wq_l[i] = (_Float16)Wq[i];
    wv_l[i] = (_Float16)Wv[i];
    wo_l[i] = (_Float16)Wo[i];          // Wo is [64][32] row-major, ld=32
  }
  if (t < SS) { bk_l[t] = bk[t]; bq_l[t] = bq[t]; bv_l[t] = bv[t]; }
  if (t < CH) { bo_l[t] = bo[t]; }

  // ---- cooperative staging: X tiles (coalesced along hw) ----------------
  // thread t -> n_local = t&7, channels c0=t>>3 and c0+32; loop over d.
  {
    const int nl = t & 7;
    const int c0 = t >> 3;
#pragma unroll
    for (int cc = 0; cc < 2; ++cc) {
      const int c = c0 + cc * 32;
      const float* xp = x + (((size_t)b * CH + c) * DD) * HW + hw0 + nl;
      _Float16* xt = (_Float16*)(smem + nl * PER_N + OFF_XT);
      float s = 0.f;
#pragma unroll 4
      for (int d = 0; d < DD; ++d) {
        float v = xp[(size_t)d * HW];
        s += v;
        xt[d * LDX + c] = (_Float16)v;   // transposed: [d][c]
      }
      xsum[nl * CH + c] = s;
    }
  }
  __syncthreads();

  // ---- per-wave attention pipeline (wave w owns position n_local = w) ---
  {
    const int wv_id = t >> 5;
    const int lane  = t & 31;
    char* nb = smem + wv_id * PER_N;
    _Float16* Xt = (_Float16*)(nb + OFF_XT);
    _Float16* kt = (_Float16*)(nb + OFF_KT);
    _Float16* qt = (_Float16*)(nb + OFF_QT);
    _Float16* vm = (_Float16*)(nb + OFF_V);
    _Float16* at = (_Float16*)(nb + OFF_AT);
    _Float16* ot = (_Float16*)(nb + OFF_OT);
    float*    sc = (float*)   (nb + OFF_SC);
    float*  ored = (float*)   (nb + OFF_ORED);
    const int mb = (lane >> 4) << 3;
    const int n16 = lane & 15;

    // K^T, Q^T stored [d][s]; V stored [s][d]
    project(wk_l, bk_l, Xt, kt, true,  lane);
    project(wq_l, bq_l, Xt, qt, true,  lane);
    project(wv_l, bv_l, Xt, vm, false, lane);

    // scores[i][j] = sum_s k[s,i] q[s,j] / sqrt(32)
    const float scale = 0.17677669529663688f;  // 1/sqrt(32)
#pragma unroll
    for (int ib = 0; ib < 2; ++ib) {
      v16h a = load_a(kt + ib * 16 * LDT, LDT, lane);   // A = K^T (M=i,K=s)
#pragma unroll
      for (int jb = 0; jb < 2; ++jb) {
        v16h bb = load_b(qt + jb * 16 * LDT, LDT, lane); // B = Q (K=s,N=j)
        v8f c = {};
        c = wmma_f16(a, bb, c);
#pragma unroll
        for (int r = 0; r < 8; ++r)
          sc[(ib * 16 + mb + r) * LDS_SC + jb * 16 + n16] = c[r] * scale;
      }
    }

    // column softmax over i (lane j = lane owns column j)
    {
      float mx = -3.0e38f;
#pragma unroll
      for (int i = 0; i < DD; ++i) mx = fmaxf(mx, sc[i * LDS_SC + lane]);
      float sum = 0.f;
#pragma unroll
      for (int i = 0; i < DD; ++i) {
        float e = __expf(sc[i * LDS_SC + lane] - mx);
        sum += e;
        sc[i * LDS_SC + lane] = e;
      }
      float inv = 1.f / sum;
#pragma unroll
      for (int i = 0; i < DD; ++i)
        at[lane * LDT + i] = (_Float16)(sc[i * LDS_SC + lane] * inv); // aT[j][i]
    }

    // O = V @ A  ->  store transposed oT[d][s]
#pragma unroll
    for (int sb = 0; sb < 2; ++sb) {
      v16h a = load_a(vm + sb * 16 * LDT, LDT, lane);   // A = V (M=s,K=d)
#pragma unroll
      for (int db = 0; db < 2; ++db) {
        v16h bb = load_b(at + db * 16 * LDT, LDT, lane); // B = A (K=i,N=j)
        v8f c = {};
        c = wmma_f16(a, bb, c);
        store_c_t_f16(ot + db * 16 * LDT + sb * 16, LDT, c, lane);
      }
    }

    // Out = Wo @ O ; fold mean-over-D via in-register N reduction
    v16h b0 = load_b(ot,            LDT, lane);
    v16h b1 = load_b(ot + 16 * LDT, LDT, lane);
#pragma unroll
    for (int cb = 0; cb < 4; ++cb) {
      v16h a = load_a(wo_l + cb * 16 * SS, SS, lane);   // Wo row-major ld=32
      v8f c0 = {}, c1 = {};
      c0 = wmma_f16(a, b0, c0);                          // N = 0..15
      c1 = wmma_f16(a, b1, c1);                          // N = 16..31
      v8f ts = c0 + c1;                                  // fold N halves
#pragma unroll
      for (int off = 1; off < 16; off <<= 1) {
#pragma unroll
        for (int r = 0; r < 8; ++r) ts[r] += __shfl_xor(ts[r], off);
      }
      if (n16 == 0) {
#pragma unroll
        for (int r = 0; r < 8; ++r) ored[cb * 16 + mb + r] = ts[r];
      }
    }
  }
  __syncthreads();

  // ---- final: out[b,c,h,w] = mean_d x + f*(mean_d(Wo@O) + bo[c]) --------
  {
    const float f = factor[0];
    const int nl = t & 7;
    const int c0 = t >> 3;
    const float* ored = (const float*)(smem + nl * PER_N + OFF_ORED);
#pragma unroll
    for (int cc = 0; cc < 2; ++cc) {
      const int c = c0 + cc * 32;
      float res = xsum[nl * CH + c] * (1.f / 32.f)
                + f * (ored[c] * (1.f / 32.f) + bo_l[c]);
      out[((size_t)b * CH + c) * HW + hw0 + nl] = res;
    }
  }
}

extern "C" void kernel_launch(void* const* d_in, const int* in_sizes, int n_in,
                              void* d_out, int out_size, void* d_ws, size_t ws_size,
                              hipStream_t stream) {
  (void)in_sizes; (void)n_in; (void)out_size; (void)d_ws; (void)ws_size;
  const float* x  = (const float*)d_in[0];
  const float* Wk = (const float*)d_in[1];
  const float* bk = (const float*)d_in[2];
  const float* Wq = (const float*)d_in[3];
  const float* bq = (const float*)d_in[4];
  const float* Wv = (const float*)d_in[5];
  const float* bv = (const float*)d_in[6];
  const float* Wo = (const float*)d_in[7];
  const float* bo = (const float*)d_in[8];
  const float* fc = (const float*)d_in[9];
  float* out = (float*)d_out;

  dim3 grid(8 * (4096 / NT));   // B * HW/NT = 4096 blocks
  dim3 block(256);              // 8 wave32s, one per position
  hipLaunchKernelGGL(attn_kernel, grid, block, SMEM_TOTAL, stream,
                     x, Wk, bk, Wq, bq, Wv, bv, Wo, bo, fc, out);
}
